// EMAVectorQuantizer_18116172055063
// MI455X (gfx1250) — compile-verified, matrix operand order checked by reading, not run on hardware
//
#include <hip/hip_runtime.h>
#include <hip/hip_bf16.h>

// ---------------------------------------------------------------------------
// EMA Vector Quantizer for MI455X (gfx1250, wave32, WMMA + TDM).
//
//  * Distance matmul on v_wmma_f32_16x16x32_bf16 with fp32->bf16 hi/lo split
//    (3 WMMAs per K-chunk: hi*hi + hi*lo + lo*hi) — ~2.6x fewer matrix ops
//    than native 16x16x4 f32 WMMA at near-fp32 accuracy.
//  * Codebook staged into LDS via tensor_load_to_lds (TDM), double-buffered,
//    TENSORcnt-synchronized; 4 waves/block share the staged tiles (4x less
//    global codebook traffic than per-wave global loads).
//  * TDM pad feature inserts 4 DWORDs every 64 DWORDs so LDS rows stride
//    272B (68 banks) -> bank-conflict-free 16-lane ds_load_b128 B-tile reads.
//  * argmin(dist) == argmax(x.e - 0.5||e||^2); strict '>' preserves
//    first-index semantics since each lane scans n ascending.
// ---------------------------------------------------------------------------

typedef __attribute__((ext_vector_type(16))) __bf16 v16bf;
typedef __attribute__((ext_vector_type(8)))  __bf16 v8bf;
typedef __attribute__((ext_vector_type(8)))  float  v8f;
typedef __attribute__((ext_vector_type(4)))  unsigned int u32x4;
typedef __attribute__((ext_vector_type(8)))  int i32x8;
typedef __attribute__((ext_vector_type(4)))  int i32x4;

#define VQ_DECAY 0.99f
#define VQ_ONE_MINUS_DECAY 0.01f
#define VQ_EPS 1e-5f

#define NB      64            // codewords staged per TDM chunk
#define ROWE    136           // LDS row stride in bf16 elems (128 + 4-DWORD pad)
#define NCHUNK  16            // 1024 / NB

// ---------------------------------------------------------------------------
// Kernel 1: split codebook into bf16 hi/lo, precompute 0.5*||e||^2, zero the
// atomic accumulators. One block per codeword, D threads.
// ---------------------------------------------------------------------------
__global__ __launch_bounds__(128)
void vq_prep(const float* __restrict__ embed,
             __bf16* __restrict__ ehi, __bf16* __restrict__ elo,
             float* __restrict__ half_norm, float* __restrict__ counts,
             float* __restrict__ esums, int D) {
    const int n = blockIdx.x;
    const int d = threadIdx.x;           // blockDim.x == D == 128
    const float e = embed[(size_t)n * D + d];
    const __bf16 h = (__bf16)e;
    const __bf16 l = (__bf16)(e - (float)h);
    ehi[(size_t)n * D + d] = h;
    elo[(size_t)n * D + d] = l;
    esums[(size_t)n * D + d] = 0.0f;     // zero accumulator every launch

    __shared__ float red[128];
    red[d] = e * e;
    __syncthreads();
    for (int s = 64; s > 0; s >>= 1) {
        if (d < s) red[d] += red[d + s];
        __syncthreads();
    }
    if (d == 0) {
        half_norm[n] = 0.5f * red[0];
        counts[n]    = 0.0f;
    }
}

// ---------------------------------------------------------------------------
// TDM 2D tile load: NB rows x 128 cols of bf16, row-padded in LDS.
// D# group0: count=1 | lds_addr | global_addr | type=2.
// D# group1: data_size=2B, pad every 64 DWORDs by 4 DWORDs,
//            tensor_dim0=tile_dim0=128, tensor_dim1=tile_dim1=NB, stride0=128.
// ---------------------------------------------------------------------------
#if __has_builtin(__builtin_amdgcn_tensor_load_to_lds)
#define VQ_HAVE_TDM 1
__device__ __forceinline__ void tdm_load_tile(const __bf16* gptr, unsigned lds_off) {
    const unsigned long long ga = (unsigned long long)(size_t)gptr;
    u32x4 g0;
    g0[0] = 1u;                                            // count=1, user mode
    g0[1] = lds_off;                                       // lds_addr (bytes)
    g0[2] = (unsigned)(ga & 0xffffffffu);                  // global_addr[31:0]
    g0[3] = (unsigned)((ga >> 32) & 0x01ffffffu)           // global_addr[56:32]
          | (2u << 30);                                    // type = 2 (image)
    i32x8 g1;
    g1[0] = (1 << 16)                                      // data_size = 2 bytes
          | (1 << 20)                                      // pad_enable
          | (5 << 22)                                      // pad_interval: 64 DW
          | (3 << 25);                                     // pad_amount: 4 DW
    g1[1] = 128 << 16;                                     // tensor_dim0 = 128
    g1[2] = NB << 16;                                      // tensor_dim1 = NB
    g1[3] = 128 << 16;                                     // tile_dim0 = 128
    g1[4] = NB;                                            // tile_dim1 = NB
    g1[5] = 128;                                           // tensor_dim0_stride
    g1[6] = 0;
    g1[7] = 0;
    const i32x4 z4 = {0, 0, 0, 0};
#if __clang_major__ >= 23
    const i32x8 z8 = {0, 0, 0, 0, 0, 0, 0, 0};
    __builtin_amdgcn_tensor_load_to_lds(g0, g1, z4, z4, z8, 0);
#else
    __builtin_amdgcn_tensor_load_to_lds(g0, g1, z4, z4, 0);
#endif
}
#else
#define VQ_HAVE_TDM 0
#endif

// ---------------------------------------------------------------------------
// Kernel 2: nearest-codeword search. 4 waves/block, each wave owns a 16-row
// A tile in registers; codebook chunks double-buffered in LDS via TDM.
// ---------------------------------------------------------------------------
__global__ __launch_bounds__(128)
void vq_assign(const float* __restrict__ x, const float* __restrict__ embed,
               const __bf16* __restrict__ ehi, const __bf16* __restrict__ elo,
               const float* __restrict__ half_norm,
               float* __restrict__ out_q,
               float* __restrict__ counts, float* __restrict__ esums,
               int N, int D) {
    const int lane = threadIdx.x & 31;
    const int wave = threadIdx.x >> 5;
    const int hlf  = lane >> 4;
    const int l16  = lane & 15;
    const int row  = (blockIdx.x * 4 + wave) * 16 + l16;

    __shared__ __align__(16) __bf16 s_hi[2][NB * ROWE];
    __shared__ __align__(16) __bf16 s_lo[2][NB * ROWE];

#if VQ_HAVE_TDM
    // Prologue: wave 0 kicks off the DMA for chunk 0 while everyone converts
    // their A tiles below (TDM overlaps with the fp32->bf16 split).
    if (wave == 0) {
        tdm_load_tile(ehi, (unsigned)(size_t)(const void*)&s_hi[0][0]);
        tdm_load_tile(elo, (unsigned)(size_t)(const void*)&s_lo[0][0]);
    }
#endif

    // --- Load + split 16 rows x 128 cols of x into WMMA A layout (bf16 hi/lo).
    // ISA 16-bit A layout: lanes 0-15 hold K=0..7 / 16..23; lanes 16-31 hold
    // K=8..15 / 24..31 per 32-wide K chunk.
    v16bf a_hi[4], a_lo[4];
    const float* xrow = x + (size_t)row * D;
#pragma unroll
    for (int kc = 0; kc < 4; ++kc) {
        const int base = kc * 32 + hlf * 8;
#pragma unroll
        for (int e = 0; e < 16; ++e) {
            const int k = base + (e < 8 ? e : (e - 8) + 16);
            const float v = xrow[k];
            const __bf16 h = (__bf16)v;
            a_hi[kc][e] = h;
            a_lo[kc][e] = (__bf16)(v - (float)h);
        }
    }

    float best[8];
    int   bidx[8];
#pragma unroll
    for (int i = 0; i < 8; ++i) { best[i] = -3.4e38f; bidx[i] = 0; }

    for (int c = 0; c < NCHUNK; ++c) {
        const int buf = c & 1;
#if VQ_HAVE_TDM
        if (wave == 0) {
            if (c + 1 < NCHUNK) {
                const size_t goff = (size_t)(c + 1) * NB * D;
                tdm_load_tile(ehi + goff,
                              (unsigned)(size_t)(const void*)&s_hi[buf ^ 1][0]);
                tdm_load_tile(elo + goff,
                              (unsigned)(size_t)(const void*)&s_lo[buf ^ 1][0]);
                __builtin_amdgcn_s_wait_tensorcnt(2);  // chunk c landed
            } else {
                __builtin_amdgcn_s_wait_tensorcnt(0);
            }
        }
#else
        // Fallback: cooperative copy with the padded row layout.
        {
            const size_t goff = (size_t)c * NB * D;
            for (int i = threadIdx.x; i < NB * D; i += 128) {
                const int r = i >> 7, cc = i & 127;
                s_hi[buf][r * ROWE + cc] = ehi[goff + i];
                s_lo[buf][r * ROWE + cc] = elo[goff + i];
            }
        }
#endif
        __syncthreads();     // chunk c visible to all 4 waves

        const __bf16* sh = &s_hi[buf][0];
        const __bf16* sl = &s_lo[buf][0];
#pragma unroll
        for (int tt = 0; tt < NB / 16; ++tt) {
            const int nloc = tt * 16 + l16;
            const int n    = c * NB + nloc;
            const __bf16* eh = sh + nloc * ROWE;
            const __bf16* el = sl + nloc * ROWE;

            v8f acc = {};
#pragma unroll
            for (int kc = 0; kc < 4; ++kc) {
                const int base = kc * 32 + hlf * 8;
                const v8bf bh0 = *(const v8bf*)(eh + base);
                const v8bf bh1 = *(const v8bf*)(eh + base + 16);
                const v8bf bl0 = *(const v8bf*)(el + base);
                const v8bf bl1 = *(const v8bf*)(el + base + 16);
                const v16bf bhi = __builtin_shufflevector(
                    bh0, bh1, 0,1,2,3,4,5,6,7,8,9,10,11,12,13,14,15);
                const v16bf blo = __builtin_shufflevector(
                    bl0, bl1, 0,1,2,3,4,5,6,7,8,9,10,11,12,13,14,15);
                acc = __builtin_amdgcn_wmma_f32_16x16x32_bf16(
                    false, a_hi[kc], false, bhi, (short)0, acc, false, false);
                acc = __builtin_amdgcn_wmma_f32_16x16x32_bf16(
                    false, a_hi[kc], false, blo, (short)0, acc, false, false);
                acc = __builtin_amdgcn_wmma_f32_16x16x32_bf16(
                    false, a_lo[kc], false, bhi, (short)0, acc, false, false);
            }

            const float adj = half_norm[n];
            // n strictly ascends per lane -> strict '>' keeps first minimum.
#pragma unroll
            for (int v = 0; v < 8; ++v) {
                const float s = acc[v] - adj;
                if (s > best[v]) { best[v] = s; bidx[v] = n; }
            }
        }
        __syncthreads();     // all waves done reading buf before overwrite
    }

    // Butterfly argmax over the 16 lanes holding different n for the same m
    // (xor masks < 16 stay inside each 16-lane half of the wave32).
#pragma unroll
    for (int m = 8; m >= 1; m >>= 1) {
#pragma unroll
        for (int v = 0; v < 8; ++v) {
            const float os = __shfl_xor(best[v], m, 32);
            const int   oi = __shfl_xor(bidx[v], m, 32);
            if (os > best[v] || (os == best[v] && oi < bidx[v])) {
                best[v] = os; bidx[v] = oi;
            }
        }
    }

    __shared__ int s_idx[64];
    if (l16 == 0) {
#pragma unroll
        for (int v = 0; v < 8; ++v) {
            const int m = v + hlf * 8;           // D layout: VGPR slot v -> row m
            s_idx[wave * 16 + m] = bidx[v];
            atomicAdd(&counts[bidx[v]], 1.0f);   // global_atomic_add_f32
        }
    }
    __syncthreads();

    // Quantized output (straight-through: x + (q - x)) and embed segment sums.
    const int t = threadIdx.x;                   // == d, since D == 128
    const size_t rowbase = (size_t)blockIdx.x * 64;
    for (int r = 0; r < 64; ++r) {
        const int ci = s_idx[r];
        const size_t g = (rowbase + r) * D + t;
        const float xv = x[g];
        const float q  = embed[(size_t)ci * D + t];
        out_q[g] = xv + (q - xv);
        atomicAdd(&esums[(size_t)ci * D + t], xv);
    }
}

// ---------------------------------------------------------------------------
// Kernel 3: EMA update + Laplace-smoothed normalization. One 1024-thread
// block (one thread per codeword), LDS tree-sum for n = sum(new_cluster_size).
// ---------------------------------------------------------------------------
__global__ __launch_bounds__(1024)
void vq_finalize(const float* __restrict__ cluster_size,
                 const float* __restrict__ embed_avg,
                 const float* __restrict__ counts,
                 const float* __restrict__ esums,
                 float* __restrict__ new_embed, float* __restrict__ new_cs,
                 float* __restrict__ new_ea, int N, int D) {
    const int i = threadIdx.x;                   // blockDim.x == N == 1024
    __shared__ float red[1024];
    const float ncs = VQ_DECAY * cluster_size[i] + VQ_ONE_MINUS_DECAY * counts[i];
    new_cs[i] = ncs;
    red[i] = ncs;
    __syncthreads();
    for (int s = 512; s > 0; s >>= 1) {
        if (i < s) red[i] += red[i + s];
        __syncthreads();
    }
    const float n  = red[0];
    const float cs = (ncs + VQ_EPS) / (n + (float)N * VQ_EPS) * n;
    for (int d = 0; d < D; ++d) {
        const size_t g = (size_t)i * D + d;
        const float nea = VQ_DECAY * embed_avg[g] + VQ_ONE_MINUS_DECAY * esums[g];
        new_ea[g]    = nea;
        new_embed[g] = nea / cs;
    }
}

// ---------------------------------------------------------------------------
extern "C" void kernel_launch(void* const* d_in, const int* in_sizes, int n_in,
                              void* d_out, int out_size, void* d_ws, size_t ws_size,
                              hipStream_t stream) {
    const float* x            = (const float*)d_in[0];
    const float* embed        = (const float*)d_in[1];
    const float* cluster_size = (const float*)d_in[2];
    const float* embed_avg    = (const float*)d_in[3];

    const int TD = in_sizes[0];      // 32*2048*128
    const int ND = in_sizes[1];      // 1024*128
    const int N  = in_sizes[2];      // 1024
    const int D  = ND / N;           // 128
    const int T  = TD / D;           // 65536

    // Workspace carve-out (everything naturally 256B-aligned).
    char* ws = (char*)d_ws;
    __bf16* ehi      = (__bf16*)ws;  ws += sizeof(__bf16) * (size_t)ND;
    __bf16* elo      = (__bf16*)ws;  ws += sizeof(__bf16) * (size_t)ND;
    float* half_norm = (float*)ws;   ws += sizeof(float) * (size_t)N;
    float* counts    = (float*)ws;   ws += sizeof(float) * (size_t)N;
    float* esums     = (float*)ws;

    // Output tuple, concatenated flat in return order.
    float* out_q     = (float*)d_out;
    float* new_embed = out_q + TD;
    float* new_cs    = new_embed + ND;
    float* new_ea    = new_cs + N;

    vq_prep<<<N, D, 0, stream>>>(embed, ehi, elo, half_norm, counts, esums, D);
    vq_assign<<<T / 64, 128, 0, stream>>>(x, embed, ehi, elo, half_norm,
                                          out_q, counts, esums, N, D);
    vq_finalize<<<1, N, 0, stream>>>(cluster_size, embed_avg, counts, esums,
                                     new_embed, new_cs, new_ea, N, D);
}
